// JointTypeHead_3272765079680
// MI455X (gfx1250) — compile-verified, matrix operand order checked by reading, not run on hardware
//
#include <hip/hip_runtime.h>
#include <stdint.h>

typedef __attribute__((ext_vector_type(16))) __bf16 v16bf;
typedef __attribute__((ext_vector_type(8)))  float  v8f;

#define DD 512
#define LN_EPS 1e-5f

__device__ __forceinline__ uint16_t f2bf(float f) {
  uint32_t u = __float_as_uint(f);
  u += 0x7FFFu + ((u >> 16) & 1u);   // round-to-nearest-even
  return (uint16_t)(u >> 16);
}

// Load one 16-bit WMMA fragment (A or B layout): lane r = row, lane-half kh
// selects K-halves; elements 0-7 = K[kh*8 .. kh*8+7], 8-15 = K[16+kh*8 ..].
__device__ __forceinline__ v16bf load_frag(const uint16_t* p) {
  union { v16bf v; float4 q[2]; } u;
  u.q[0] = *(const float4*)(p);
  u.q[1] = *(const float4*)(p + 16);
  return u.v;
}

// ---------------- small prep kernels ----------------
__global__ void f32_to_bf16_kernel(const float* __restrict__ in,
                                   uint16_t* __restrict__ out, int n) {
  int i = blockIdx.x * blockDim.x + threadIdx.x;
  if (i < n) out[i] = f2bf(in[i]);
}

__global__ void build_wout_kernel(const float* __restrict__ in,
                                  uint16_t* __restrict__ out, int nvalid) {
  int i = blockIdx.x * blockDim.x + threadIdx.x;
  if (i >= 16 * DD) return;
  out[i] = (i < nvalid) ? f2bf(in[i]) : (uint16_t)0;
}

__global__ void zero_kernel(float* p, int n) {
  int i = blockIdx.x * blockDim.x + threadIdx.x;
  if (i < n) p[i] = 0.f;
}

// ---------------- edge gather + LayerNorm -> bf16 ----------------
__global__ __launch_bounds__(128)
void edge_ln_kernel(const float* __restrict__ x, const int* __restrict__ src,
                    const int* __restrict__ dst, const float* __restrict__ lnw,
                    uint16_t* __restrict__ out, int E) {
  const int e = blockIdx.x;
  const int t = threadIdx.x;
  uint16_t* orow = out + (size_t)e * DD + t * 4;
  if (e >= E) {                       // pad rows: exact zeros (keeps BN sums exact)
    uint2 z; z.x = 0u; z.y = 0u;
    *(uint2*)orow = z;
    return;
  }
  const float4 a = *(const float4*)(x + (size_t)src[e] * DD + t * 4);
  const float4 b = *(const float4*)(x + (size_t)dst[e] * DD + t * 4);
  float h0 = a.x + b.x, h1 = a.y + b.y, h2 = a.z + b.z, h3 = a.w + b.w;
  float ps  = h0 + h1 + h2 + h3;
  float pss = h0 * h0 + h1 * h1 + h2 * h2 + h3 * h3;
  for (int o = 16; o > 0; o >>= 1) {  // wave32 reduction
    ps  += __shfl_xor(ps,  o, 32);
    pss += __shfl_xor(pss, o, 32);
  }
  __shared__ float red[2][4];
  const int w = t >> 5, l = t & 31;
  if (l == 0) { red[0][w] = ps; red[1][w] = pss; }
  __syncthreads();
  const float S  = red[0][0] + red[0][1] + red[0][2] + red[0][3];
  const float SS = red[1][0] + red[1][1] + red[1][2] + red[1][3];
  const float mu   = S * (1.f / DD);
  const float var  = SS * (1.f / DD) - mu * mu;
  const float rstd = rsqrtf(var + LN_EPS);
  const float4 g = *(const float4*)(lnw + t * 4);
  uint2 pk;
  pk.x = (uint32_t)f2bf((h0 - mu) * rstd * g.x) |
         ((uint32_t)f2bf((h1 - mu) * rstd * g.y) << 16);
  pk.y = (uint32_t)f2bf((h2 - mu) * rstd * g.z) |
         ((uint32_t)f2bf((h3 - mu) * rstd * g.w) << 16);
  *(uint2*)orow = pk;
}

// ---------------- GEMM: Y = A(bf16) @ W(bf16)^T, + BN column stats ----------------
// Block: 128x128 tile, 8 waves (4 in M x 2 in N), each wave 32x64 = 2x4 WMMA tiles.
__global__ __launch_bounds__(256)
void gemm_bn_stats_kernel(const uint16_t* __restrict__ A,   // [Epad, DD] bf16
                          const uint16_t* __restrict__ B,   // [DD, DD] bf16 (row n = W[n,:])
                          float* __restrict__ Y,            // [Epad, DD] f32
                          float* __restrict__ colsum,       // [DD]
                          float* __restrict__ colsumsq) {   // [DD]
  const int lane  = threadIdx.x & 31;
  const int wave  = threadIdx.x >> 5;
  const int waveM = wave >> 1;           // 0..3
  const int waveN = wave & 1;            // 0..1
  const int bm = blockIdx.x * 128;
  const int bn = blockIdx.y * 128;
  const int r  = lane & 15;
  const int kh = lane >> 4;

  v8f acc[2][4] = {};

  const uint16_t* Abase = A + (size_t)(bm + waveM * 32 + r) * DD + kh * 8;
  const uint16_t* Bbase = B + (size_t)(bn + waveN * 64 + r) * DD + kh * 8;

#pragma unroll 2
  for (int k = 0; k < DD; k += 32) {
    v16bf af[2], bf[4];
#pragma unroll
    for (int tm = 0; tm < 2; ++tm) af[tm] = load_frag(Abase + (size_t)tm * 16 * DD + k);
#pragma unroll
    for (int tn = 0; tn < 4; ++tn) bf[tn] = load_frag(Bbase + (size_t)tn * 16 * DD + k);
#pragma unroll
    for (int tm = 0; tm < 2; ++tm)
#pragma unroll
      for (int tn = 0; tn < 4; ++tn)
        acc[tm][tn] = __builtin_amdgcn_wmma_f32_16x16x32_bf16(
            false, af[tm], false, bf[tn], (short)0, acc[tm][tn], false, false);
  }

  // store Y (f32): D layout -> lane col = r, rows = v + kh*8
#pragma unroll
  for (int tm = 0; tm < 2; ++tm) {
    const int row0 = bm + waveM * 32 + tm * 16 + kh * 8;
#pragma unroll
    for (int tn = 0; tn < 4; ++tn) {
      const int col = bn + waveN * 64 + tn * 16 + r;
      float* yp = Y + (size_t)row0 * DD + col;
#pragma unroll
      for (int v = 0; v < 8; ++v) yp[(size_t)v * DD] = acc[tm][tn][v];
    }
  }

  // BN column statistics: reduce 32 rows per wave-tile-column, then atomics
#pragma unroll
  for (int tn = 0; tn < 4; ++tn) {
    float s = 0.f, ss = 0.f;
#pragma unroll
    for (int tm = 0; tm < 2; ++tm)
#pragma unroll
      for (int v = 0; v < 8; ++v) { float t = acc[tm][tn][v]; s += t; ss += t * t; }
    s  += __shfl_xor(s,  16, 32);    // combine lane halves (rows 0-7 with 8-15)
    ss += __shfl_xor(ss, 16, 32);
    if (lane < 16) {
      const int col = bn + waveN * 64 + tn * 16 + lane;
      atomicAdd(&colsum[col], s);
      atomicAdd(&colsumsq[col], ss);
    }
  }
}

// ---------------- BN finalize: fold mean/var/gamma/beta into scale/bias ----------------
__global__ void bn_finalize_kernel(const float* __restrict__ sum,
                                   const float* __restrict__ ssq,
                                   const float* __restrict__ g,
                                   const float* __restrict__ b,
                                   float* __restrict__ scale,
                                   float* __restrict__ bias, float invE) {
  int c = blockIdx.x * blockDim.x + threadIdx.x;
  if (c >= DD) return;
  const float m   = sum[c] * invE;
  const float var = ssq[c] * invE - m * m;
  const float rs  = rsqrtf(var + LN_EPS);
  const float sc  = g[c] * rs;
  scale[c] = sc;
  bias[c]  = fmaf(-m, sc, b[c]);
}

// ---------------- BN apply + ReLU -> bf16 ----------------
__global__ __launch_bounds__(256)
void bn_relu_kernel(const float* __restrict__ Y, const float* __restrict__ scale,
                    const float* __restrict__ bias, uint16_t* __restrict__ out,
                    int E, long long total) {
  const long long t = (long long)blockIdx.x * 256 + threadIdx.x;
  if (t >= total) return;
  const int row = (int)(t >> 7);
  const int c   = (int)(t & 127) << 2;
  uint2 pk;
  if (row >= E) {
    pk.x = 0u; pk.y = 0u;            // keep pad rows exactly zero for next layer
  } else {
    const float4 y = *(const float4*)(Y + (size_t)row * DD + c);
    const float4 s = *(const float4*)(scale + c);
    const float4 b = *(const float4*)(bias + c);
    const float v0 = fmaxf(fmaf(y.x, s.x, b.x), 0.f);
    const float v1 = fmaxf(fmaf(y.y, s.y, b.y), 0.f);
    const float v2 = fmaxf(fmaf(y.z, s.z, b.z), 0.f);
    const float v3 = fmaxf(fmaf(y.w, s.w, b.w), 0.f);
    pk.x = (uint32_t)f2bf(v0) | ((uint32_t)f2bf(v1) << 16);
    pk.y = (uint32_t)f2bf(v2) | ((uint32_t)f2bf(v3) << 16);
  }
  *(uint2*)(out + (size_t)row * DD + c) = pk;
}

// ---------------- output head: logits = h @ Wout^T (N padded to 16) ----------------
__global__ __launch_bounds__(256)
void out_gemm_kernel(const uint16_t* __restrict__ A, const uint16_t* __restrict__ Bw,
                     float* __restrict__ out, int E, int NC) {
  const int lane = threadIdx.x & 31;
  const int wave = threadIdx.x >> 5;
  const int r  = lane & 15;
  const int kh = lane >> 4;
  const int r0 = (blockIdx.x * 8 + wave) * 16;
  v8f acc = {};
  const uint16_t* Ab = A  + (size_t)(r0 + r) * DD + kh * 8;
  const uint16_t* Bb = Bw + (size_t)r * DD + kh * 8;
#pragma unroll 4
  for (int k = 0; k < DD; k += 32) {
    v16bf a = load_frag(Ab + k);
    v16bf b = load_frag(Bb + k);
    acc = __builtin_amdgcn_wmma_f32_16x16x32_bf16(false, a, false, b, (short)0,
                                                  acc, false, false);
  }
  if (r < NC) {
#pragma unroll
    for (int v = 0; v < 8; ++v) {
      const int row = r0 + v + kh * 8;
      if (row < E) out[(size_t)row * NC + r] = acc[v];
    }
  }
}

extern "C" void kernel_launch(void* const* d_in, const int* in_sizes, int n_in,
                              void* d_out, int out_size, void* d_ws, size_t ws_size,
                              hipStream_t stream) {
  const float* x    = (const float*)d_in[0];
  const int*   src  = (const int*)d_in[1];
  const int*   dst  = (const int*)d_in[2];
  const float* lnw  = (const float*)d_in[3];
  const float* W1   = (const float*)d_in[4];
  const float* g1   = (const float*)d_in[5];
  const float* b1   = (const float*)d_in[6];
  const float* W2   = (const float*)d_in[7];
  const float* g2   = (const float*)d_in[8];
  const float* b2   = (const float*)d_in[9];
  const float* Wout = (const float*)d_in[10];
  float* out = (float*)d_out;

  const int E    = in_sizes[1];
  const int NC   = in_sizes[10] / DD;
  const int Epad = ((E + 127) / 128) * 128;

  // workspace layout
  char* p = (char*)d_ws;
  uint16_t* hA    = (uint16_t*)p; p += (size_t)Epad * DD * 2;   // bf16 activations
  float*    Y     = (float*)p;    p += (size_t)Epad * DD * 4;   // pre-BN f32
  uint16_t* W1b   = (uint16_t*)p; p += (size_t)DD * DD * 2;
  uint16_t* W2b   = (uint16_t*)p; p += (size_t)DD * DD * 2;
  uint16_t* Woutb = (uint16_t*)p; p += (size_t)16 * DD * 2;     // padded to 16 rows
  float* sum1 = (float*)p; p += DD * 4;
  float* ssq1 = (float*)p; p += DD * 4;
  float* sc1  = (float*)p; p += DD * 4;
  float* bi1  = (float*)p; p += DD * 4;
  float* sum2 = (float*)p; p += DD * 4;
  float* ssq2 = (float*)p; p += DD * 4;
  float* sc2  = (float*)p; p += DD * 4;
  float* bi2  = (float*)p; p += DD * 4;
  (void)ws_size; (void)n_in; (void)out_size;

  const int nW = DD * DD;
  f32_to_bf16_kernel<<<(nW + 255) / 256, 256, 0, stream>>>(W1, W1b, nW);
  f32_to_bf16_kernel<<<(nW + 255) / 256, 256, 0, stream>>>(W2, W2b, nW);
  build_wout_kernel<<<(16 * DD + 255) / 256, 256, 0, stream>>>(Wout, Woutb, NC * DD);
  zero_kernel<<<(2 * DD + 255) / 256, 256, 0, stream>>>(sum1, 2 * DD); // sum1+ssq1
  zero_kernel<<<(2 * DD + 255) / 256, 256, 0, stream>>>(sum2, 2 * DD); // sum2+ssq2

  edge_ln_kernel<<<Epad, 128, 0, stream>>>(x, src, dst, lnw, hA, E);

  const dim3 gg(Epad / 128, DD / 128);
  const long long total = (long long)Epad * (DD / 4);
  const unsigned bnBlocks = (unsigned)((total + 255) / 256);
  const float invE = 1.f / (float)E;

  gemm_bn_stats_kernel<<<gg, 256, 0, stream>>>(hA, W1b, Y, sum1, ssq1);
  bn_finalize_kernel<<<2, 256, 0, stream>>>(sum1, ssq1, g1, b1, sc1, bi1, invE);
  bn_relu_kernel<<<bnBlocks, 256, 0, stream>>>(Y, sc1, bi1, hA, E, total);

  gemm_bn_stats_kernel<<<gg, 256, 0, stream>>>(hA, W2b, Y, sum2, ssq2);
  bn_finalize_kernel<<<2, 256, 0, stream>>>(sum2, ssq2, g2, b2, sc2, bi2, invE);
  bn_relu_kernel<<<bnBlocks, 256, 0, stream>>>(Y, sc2, bi2, hA, E, total);

  out_gemm_kernel<<<Epad / 128, 256, 0, stream>>>(hA, Woutb, out, E, NC);
}
